// NystromAttention_4638564680050
// MI455X (gfx1250) — matrix operand order repaired and checked by reading, not compiled, not run on hardware
//
#include <hip/hip_runtime.h>
#include <hip/hip_bf16.h>

// ---------------------------------------------------------------------------
// Nystrom attention for MI455X (gfx1250, wave32, WMMA 16x16x32 f16->f32)
// + async global->LDS staging (ASYNCcnt path) for the conv epilogue
// ---------------------------------------------------------------------------

typedef _Float16 h16;
typedef __attribute__((ext_vector_type(16))) _Float16 v16h;
typedef __attribute__((ext_vector_type(8)))  _Float16 v8h;
typedef __attribute__((ext_vector_type(8)))  float    v8f;

#define HEADS 8
#define DHEAD 64
#define NLM   256
#define SEQ   4096
#define BATCH 8
#define DIM   512
#define BH    (BATCH*HEADS)
#define RESK  33

// ---- WMMA fragment helpers (contiguous k = 16*(lane/16)+s in A and B) ---------------
// A (16x32 f16, row-major source, stride lda):
//   lanes 0-15: K = 0..15; lanes 16-31: K = 16..31 (matches B fragment)
__device__ __forceinline__ v16h load_a_frag(const h16* A, int lda, int lane, int k0) {
    int r  = lane & 15;
    int hf = lane >> 4;
    const h16* p = A + (long)r * lda + k0 + hf * 16;
    return *(const v16h*)(p);
}
// B from BT (row-major N x K, stride ldb): lane n = lane&15,
//   lanes 0-15 hold K=k0..k0+15, lanes 16-31 hold K=k0+16..k0+31 (one 32B load)
__device__ __forceinline__ v16h load_b_frag(const h16* BT, int ldb, int lane, int k0) {
    int r  = lane & 15;
    int hf = lane >> 4;
    const h16* p = BT + (long)r * ldb + k0 + hf * 16;
    return *(const v16h*)(p);
}
__device__ __forceinline__ v8f wmma_f16(v16h a, v16h b, v8f c) {
    return __builtin_amdgcn_wmma_f32_16x16x32_f16(false, a, false, b, (short)0, c, false, false);
}
#define ACC_ZERO {0.f,0.f,0.f,0.f,0.f,0.f,0.f,0.f}

// ---- CDNA5 async global->LDS (tracked with ASYNCcnt) -----------------------
// LDS offset operand: low 32 bits of a generic pointer to __shared__ data are
// the wave-relative LDS byte address (ISA 10.2 aperture mapping).
__device__ __forceinline__ void async_load_b128(unsigned lds_off, const void* gaddr) {
    asm volatile("global_load_async_to_lds_b128 %0, %1, off"
                 :: "v"(lds_off), "v"(gaddr) : "memory");
}
__device__ __forceinline__ void wait_async0() {
    asm volatile("s_wait_asynccnt 0" ::: "memory");
}

// ---------------------------------------------------------------------------
// 0) tiny init: clear atomic-max scalars
__global__ void k_init_scalars(unsigned* sc) { if (threadIdx.x < 8) sc[threadIdx.x] = 0u; }

// 1) conversions
__global__ void k_convert_x(const float* __restrict__ x, h16* __restrict__ xh, long n) {
    long i = (long)blockIdx.x * blockDim.x + threadIdx.x;
    if (i < n) xh[i] = (h16)x[i];
}
__global__ void k_convert_wT(const float* __restrict__ w, h16* __restrict__ wT) {
    long i = (long)blockIdx.x * blockDim.x + threadIdx.x;  // 1536*512
    if (i < (long)3 * DIM * DIM) {
        int j = (int)(i >> 9), kk = (int)(i & 511);
        wT[i] = (h16)w[(long)kk * (3 * DIM) + j];
    }
}

// 2) QKV GEMM: [32768,512]f16 @ [512,1536] via BT=[1536,512]f16, scatter epilogue
__global__ __launch_bounds__(256) void k_qkv_gemm(
    const h16* __restrict__ xh, const h16* __restrict__ wT,
    h16* __restrict__ qh, h16* __restrict__ kh,
    h16* __restrict__ vh, h16* __restrict__ vTh)
{
    int wave = threadIdx.x >> 5, lane = threadIdx.x & 31;
    int r0 = blockIdx.x * 128 + wave * 16;   // row in [0, 32768)
    int c0 = blockIdx.y * 64;                // col in [0, 1536)
    const h16* Ab = xh + (long)r0 * DIM;
    v8f acc0 = ACC_ZERO, acc1 = ACC_ZERO, acc2 = ACC_ZERO, acc3 = ACC_ZERO;
#pragma unroll 4
    for (int k0 = 0; k0 < DIM; k0 += 32) {
        if (k0 + 64 < DIM)
            __builtin_prefetch(Ab + (long)(lane & 15) * DIM + k0 + 64, 0, 1); // global_prefetch
        v16h a = load_a_frag(Ab, DIM, lane, k0);
        acc0 = wmma_f16(a, load_b_frag(wT + (long)(c0 +  0) * DIM, DIM, lane, k0), acc0);
        acc1 = wmma_f16(a, load_b_frag(wT + (long)(c0 + 16) * DIM, DIM, lane, k0), acc1);
        acc2 = wmma_f16(a, load_b_frag(wT + (long)(c0 + 32) * DIM, DIM, lane, k0), acc2);
        acc3 = wmma_f16(a, load_b_frag(wT + (long)(c0 + 48) * DIM, DIM, lane, k0), acc3);
    }
    int cl = lane & 15, rb = (lane >> 4) * 8;
    v8f accs[4] = {acc0, acc1, acc2, acc3};
#pragma unroll
    for (int t = 0; t < 4; ++t) {
#pragma unroll
        for (int g = 0; g < 8; ++g) {
            int row = r0 + rb + g;
            int col = c0 + t * 16 + cl;
            int bb = row >> 12, nn = row & 4095;
            int part = col >> 9, hh = (col >> 6) & 7, dd = col & 63;
            long base = (((long)(bb * 8 + hh) * SEQ) + nn) * DHEAD + dd;
            float v = accs[t][g];
            if (part == 0)      qh[base] = (h16)(v * 0.125f);     // q * d^-0.5
            else if (part == 1) kh[base] = (h16)v;
            else {
                vh[base] = (h16)v;
                vTh[((long)(bb * 8 + hh) * DHEAD + dd) * SEQ + nn] = (h16)v;
            }
        }
    }
}

// 3) landmark means: grid = BH*NLM blocks, 64 threads
__global__ __launch_bounds__(64) void k_landmark_means(
    const h16* __restrict__ qh, const h16* __restrict__ kh,
    h16* __restrict__ qlh, h16* __restrict__ klh)
{
    long idx = blockIdx.x;            // bh*256 + mm
    int  c   = threadIdx.x;
    long base = idx * 16 * DHEAD;
    float sq = 0.f, sk = 0.f;
#pragma unroll
    for (int t = 0; t < 16; ++t) {
        sq += (float)qh[base + t * DHEAD + c];
        sk += (float)kh[base + t * DHEAD + c];
    }
    qlh[idx * DHEAD + c] = (h16)(sq * 0.0625f);
    klh[idx * DHEAD + c] = (h16)(sk * 0.0625f);
}

// 4) generic batched WMMA GEMM: S = alpha*I + scale*(A@B), B given transposed.
//    Optional outputs: f32 row-major, f16 row-major, f16 transposed.
__global__ __launch_bounds__(32) void k_gemm_f16(
    float* outF, h16* outH, h16* outTH,
    const h16* __restrict__ A, const h16* __restrict__ BT,
    int K, int lda, int ldb, int ldc, int ldct,
    long sA, long sB, long sC, long sCT,
    float alpha, float scale)
{
    int batch = blockIdx.x, tm = blockIdx.y, tn = blockIdx.z;
    int lane = threadIdx.x & 31;
    const h16* Ab = A  + (long)batch * sA + (long)(tm * 16) * lda;
    const h16* Bb = BT + (long)batch * sB + (long)(tn * 16) * ldb;
    v8f acc = ACC_ZERO;
    for (int k0 = 0; k0 < K; k0 += 32)
        acc = wmma_f16(load_a_frag(Ab, lda, lane, k0),
                       load_b_frag(Bb, ldb, lane, k0), acc);
    int cl = lane & 15, rb = (lane >> 4) * 8;
#pragma unroll
    for (int g = 0; g < 8; ++g) {
        int row = tm * 16 + rb + g;
        int col = tn * 16 + cl;
        float v = scale * acc[g] + ((row == col) ? alpha : 0.f);
        if (outF)  outF [(long)batch * sC  + (long)row * ldc  + col] = v;
        if (outH)  outH [(long)batch * sC  + (long)row * ldc  + col] = (h16)v;
        if (outTH) outTH[(long)batch * sCT + (long)col * ldct + row] = (h16)v;
    }
}

// 5) row softmax for attn2 (rows of 256), writes f32 in-place + f16 copy
__global__ __launch_bounds__(256) void k_softmax256(
    float* __restrict__ X, h16* __restrict__ Xh)
{
    __shared__ float sh[256];
    long row = blockIdx.x;
    int tid = threadIdx.x;
    float v = X[row * 256 + tid];
    sh[tid] = v; __syncthreads();
    for (int s = 128; s > 0; s >>= 1) { if (tid < s) sh[tid] = fmaxf(sh[tid], sh[tid + s]); __syncthreads(); }
    float mx = sh[0]; __syncthreads();
    float e = __expf(v - mx);
    sh[tid] = e; __syncthreads();
    for (int s = 128; s > 0; s >>= 1) { if (tid < s) sh[tid] += sh[tid + s]; __syncthreads(); }
    float r = e / sh[0];
    X [row * 256 + tid] = r;
    Xh[row * 256 + tid] = (h16)r;
}

// 6) global max of |row sums| and |col sums| (pinv scaling); uint-bit atomicMax (values >= 0)
__global__ __launch_bounds__(256) void k_absmax(const float* __restrict__ X, unsigned* sc)
{
    __shared__ float s1[256], s2[256];
    long bh = blockIdx.x >> 8; int i = blockIdx.x & 255;
    int tid = threadIdx.x;
    const float* Xb = X + bh * 65536;
    s1[tid] = fabsf(Xb[(long)i * 256 + tid]);   // row i   (col = sum axis -1)
    s2[tid] = fabsf(Xb[(long)tid * 256 + i]);   // col i   (row = sum axis -2)
    __syncthreads();
    for (int s = 128; s > 0; s >>= 1) {
        if (tid < s) { s1[tid] += s1[tid + s]; s2[tid] += s2[tid + s]; }
        __syncthreads();
    }
    if (tid == 0) {
        atomicMax(&sc[0], __float_as_uint(s1[0]));
        atomicMax(&sc[1], __float_as_uint(s2[0]));
    }
}

// 7) z0 = X^T / (cmax*rmax): write zh and zTh
__global__ __launch_bounds__(256) void k_zinit(
    const float* __restrict__ X, const unsigned* __restrict__ sc,
    h16* __restrict__ zh, h16* __restrict__ zTh)
{
    float inv = 1.0f / (__uint_as_float(sc[0]) * __uint_as_float(sc[1]));
    long idx = (long)blockIdx.x * blockDim.x + threadIdx.x;
    if (idx < (long)BH * 65536) {
        long bh = idx >> 16; int r = (int)(idx >> 8) & 255, c = (int)idx & 255;
        const float* Xb = X + bh * 65536;
        zh [idx] = (h16)(Xb[(long)c * 256 + r] * inv);
        zTh[idx] = (h16)(Xb[(long)r * 256 + c] * inv);
    }
}

// 8) outT[r][c] = alpha*I - A^T  (i.e. T1^T = 7I - A1^T)
__global__ __launch_bounds__(256) void k_aI_minus_AT(
    const h16* __restrict__ Ain, h16* __restrict__ outT, float alpha)
{
    long idx = (long)blockIdx.x * blockDim.x + threadIdx.x;
    if (idx < (long)BH * 65536) {
        long bh = idx >> 16; int r = (int)(idx >> 8) & 255, c = (int)idx & 255;
        float v = ((r == c) ? alpha : 0.f) - (float)Ain[bh * 65536 + (long)c * 256 + r];
        outT[idx] = (h16)v;
    }
}

// 9) fused sim3 -> softmax(axis=n) -> attn3 @ V.  One block = 16 landmark rows.
//    16x4096 f16 logit block lives in LDS (128KB of the 320KB WGP LDS).
__global__ __launch_bounds__(256) void k_sim3_softmax_pv(
    const h16* __restrict__ qlh, const h16* __restrict__ kh,
    const h16* __restrict__ vTh, float* __restrict__ attn3v)
{
    extern __shared__ char smem_dyn[];
    h16* S = (h16*)smem_dyn;                  // [16][4096]
    int bh = blockIdx.x, rb = blockIdx.y;     // 64 x 16
    int wave = threadIdx.x >> 5, lane = threadIdx.x & 31;
    const h16* Aq = qlh + ((long)bh * NLM + rb * 16) * DHEAD;
    const h16* Bk = kh  + (long)bh * SEQ * DHEAD;

    // sim3 tile row: 256 N-tiles over waves
    for (int t = wave; t < 256; t += 8) {
        v8f acc = ACC_ZERO;
        acc = wmma_f16(load_a_frag(Aq, DHEAD, lane, 0),
                       load_b_frag(Bk + (long)t * 16 * DHEAD, DHEAD, lane, 0), acc);
        acc = wmma_f16(load_a_frag(Aq, DHEAD, lane, 32),
                       load_b_frag(Bk + (long)t * 16 * DHEAD, DHEAD, lane, 32), acc);
        int cl = lane & 15, rg = (lane >> 4) * 8;
#pragma unroll
        for (int g = 0; g < 8; ++g) S[(long)(rg + g) * SEQ + t * 16 + cl] = (h16)acc[g];
    }
    __syncthreads();

    // softmax over n: wave handles rows wave and wave+8 (wave32 shuffles)
#pragma unroll
    for (int rr = 0; rr < 2; ++rr) {
        int row = wave + rr * 8;
        h16* Sr = S + (long)row * SEQ;
        float mx = -3.0e38f;
        for (int i = lane; i < SEQ; i += 32) mx = fmaxf(mx, (float)Sr[i]);
        for (int o = 16; o; o >>= 1) mx = fmaxf(mx, __shfl_xor(mx, o, 32));
        float sum = 0.f;
        for (int i = lane; i < SEQ; i += 32) sum += __expf((float)Sr[i] - mx);
        for (int o = 16; o; o >>= 1) sum += __shfl_xor(sum, o, 32);
        float rs = 1.0f / sum;
        for (int i = lane; i < SEQ; i += 32) Sr[i] = (h16)(__expf((float)Sr[i] - mx) * rs);
    }
    __syncthreads();

    // attn3 @ v : waves 0..3 each own one 16x16 d-tile, K = 4096 from LDS
    if (wave < 4) {
        v8f acc = ACC_ZERO;
        const h16* Bv = vTh + (long)bh * DHEAD * SEQ + (long)wave * 16 * SEQ;
        for (int k0 = 0; k0 < SEQ; k0 += 32)
            acc = wmma_f16(load_a_frag(S, SEQ, lane, k0),
                           load_b_frag(Bv, SEQ, lane, k0), acc);
        float* C = attn3v + ((long)bh * NLM + rb * 16) * DHEAD + wave * 16;
        int cl = lane & 15, rg = (lane >> 4) * 8;
#pragma unroll
        for (int g = 0; g < 8; ++g) C[(long)(rg + g) * DHEAD + cl] = acc[g];
    }
}

// 10) attn3v [m,d]f32 -> transposed f16 [d,m]
__global__ __launch_bounds__(256) void k_a3v_transpose(
    const float* __restrict__ a3v, h16* __restrict__ a3vT)
{
    long idx = (long)blockIdx.x * blockDim.x + threadIdx.x;
    if (idx < (long)BH * DHEAD * NLM) {
        long bh = idx >> 14; int dd = (int)(idx >> 8) & 63, mm = (int)idx & 255;
        a3vT[idx] = (h16)a3v[bh * (NLM * DHEAD) + (long)mm * DHEAD + dd];
    }
}

// 11) fused sim1 -> softmax -> @W -> depthwise residual conv -> partial max.
//     The 96x64 V window for the conv is DMA'd to LDS at kernel entry via
//     global_load_async_to_lds_b128 and consumed after s_wait_asynccnt 0.
__global__ __launch_bounds__(256) void k_sim1_softmax_out(
    const h16* __restrict__ qh, const h16* __restrict__ klh,
    const h16* __restrict__ WTh, const h16* __restrict__ vh,
    const float* __restrict__ w_res, float* __restrict__ partials)
{
    __shared__ h16  P[64 * 256];               // 32KB softmax block
    __shared__ float O[64 * 64];               // 16KB output block
    __shared__ float red[4 * 64];
    __shared__ __align__(16) h16 Vs[96 * 64];  // 12KB async-staged V window
    int bh = blockIdx.x, nb = blockIdx.y;      // 64 x 64
    int hh = bh & 7;
    int wave = threadIdx.x >> 5, lane = threadIdx.x & 31;
    int n0 = nb * 64;
    const h16* Aq  = qh  + ((long)bh * SEQ + n0) * DHEAD;
    const h16* Bkl = klh + (long)bh * NLM * DHEAD;
    const h16* Vb  = vh  + (long)bh * SEQ * DHEAD;

    // kick off async DMA of V rows [n0-16, n0+80) x 64 (768 x b128 chunks);
    // halo rows are zero-filled with plain DS stores (disjoint LDS addresses,
    // so no async-vs-DS ordering hazard)
    {
        unsigned ldsbase = (unsigned)(unsigned long long)(const void*)Vs;
        for (int c = threadIdx.x; c < 768; c += 256) {
            int l  = c >> 3;                    // window row 0..95
            int nn = n0 - 16 + l;
            if (nn >= 0 && nn < SEQ) {
                async_load_b128(ldsbase + c * 16,
                                (const void*)(Vb + (long)nn * DHEAD + (c & 7) * 8));
            } else {
                uint4 z{0u, 0u, 0u, 0u};
                *(uint4*)((char*)(void*)Vs + c * 16) = z;
            }
        }
    }

    // sim1 block [64 x 256]: 64 tiles over 8 waves (overlaps the async DMA)
    for (int t = wave; t < 64; t += 8) {
        int tr = t >> 4, tc = t & 15;
        v8f acc = ACC_ZERO;
        acc = wmma_f16(load_a_frag(Aq + (long)tr * 16 * DHEAD, DHEAD, lane, 0),
                       load_b_frag(Bkl + (long)tc * 16 * DHEAD, DHEAD, lane, 0), acc);
        acc = wmma_f16(load_a_frag(Aq + (long)tr * 16 * DHEAD, DHEAD, lane, 32),
                       load_b_frag(Bkl + (long)tc * 16 * DHEAD, DHEAD, lane, 32), acc);
        int cl = lane & 15, rg = (lane >> 4) * 8;
#pragma unroll
        for (int g = 0; g < 8; ++g) P[(tr * 16 + rg + g) * 256 + tc * 16 + cl] = (h16)acc[g];
    }
    __syncthreads();

    // softmax rows (64 rows, 8 per wave)
    for (int r = wave; r < 64; r += 8) {
        h16* Pr = P + r * 256;
        float mx = -3.0e38f;
        for (int i = lane; i < 256; i += 32) mx = fmaxf(mx, (float)Pr[i]);
        for (int o = 16; o; o >>= 1) mx = fmaxf(mx, __shfl_xor(mx, o, 32));
        float sum = 0.f;
        for (int i = lane; i < 256; i += 32) sum += __expf((float)Pr[i] - mx);
        for (int o = 16; o; o >>= 1) sum += __shfl_xor(sum, o, 32);
        float rs = 1.0f / sum;
        for (int i = lane; i < 256; i += 32) Pr[i] = (h16)(__expf((float)Pr[i] - mx) * rs);
    }
    __syncthreads();

    // O = P @ W : 16 tiles over 8 waves, K=256, W supplied transposed f16
    for (int t = wave; t < 16; t += 8) {
        int tr = t >> 2, tc = t & 3;
        v8f acc = ACC_ZERO;
        const h16* Bw = WTh + (long)bh * DHEAD * NLM + (long)tc * 16 * NLM;
        for (int k0 = 0; k0 < 256; k0 += 32)
            acc = wmma_f16(load_a_frag(P + tr * 16 * 256, 256, lane, k0),
                           load_b_frag(Bw, 256, lane, k0), acc);
        int cl = lane & 15, rg = (lane >> 4) * 8;
#pragma unroll
        for (int g = 0; g < 8; ++g) O[(tr * 16 + rg + g) * 64 + tc * 16 + cl] = acc[g];
    }
    wait_async0();       // per-wave: all our async LDS writes have landed
    __syncthreads();     // publish to the whole workgroup

    // residual depthwise conv (kernel 33 over sequence) from LDS + block max
    int c = threadIdx.x & 63, rg4 = threadIdx.x >> 6;
    float mx = -3.0e38f;
    for (int rr = 0; rr < 16; ++rr) {
        int row = rg4 * 16 + rr;
        float acc = O[row * 64 + c];
#pragma unroll
        for (int j = 0; j < RESK; ++j)
            acc += w_res[hh * RESK + j] * (float)Vs[(row + j) * 64 + c];
        mx = fmaxf(mx, acc);
    }
    red[rg4 * 64 + c] = mx;
    __syncthreads();
    if (threadIdx.x < 64) {
        float m = fmaxf(fmaxf(red[threadIdx.x], red[64 + threadIdx.x]),
                        fmaxf(red[128 + threadIdx.x], red[192 + threadIdx.x]));
        partials[((long)bh * 64 + nb) * 64 + threadIdx.x] = m;
    }
}

// 12) final: max over partial blocks, then [64] @ w_out[64,512] + b_out
__global__ __launch_bounds__(512) void k_final_proj(
    const float* __restrict__ partials, const float* __restrict__ w_out,
    const float* __restrict__ b_out, float* __restrict__ out)
{
    __shared__ float mx[64];
    int bh = blockIdx.x;
    if (threadIdx.x < 64) {
        float m = -3.0e38f;
        for (int k = 0; k < 64; ++k)
            m = fmaxf(m, partials[((long)bh * 64 + k) * 64 + threadIdx.x]);
        mx[threadIdx.x] = m;
    }
    __syncthreads();
    int j = threadIdx.x;
    float s = b_out[j];
    for (int c = 0; c < 64; ++c) s += mx[c] * w_out[(long)c * DIM + j];
    out[(long)bh * DIM + j] = s;
}

// ---------------------------------------------------------------------------
extern "C" void kernel_launch(void* const* d_in, const int* in_sizes, int n_in,
                              void* d_out, int out_size, void* d_ws, size_t ws_size,
                              hipStream_t stream) {
    (void)in_sizes; (void)n_in; (void)out_size; (void)ws_size;
    const float* x     = (const float*)d_in[0];
    const float* w_qkv = (const float*)d_in[1];
    const float* w_out = (const float*)d_in[2];
    const float* b_out = (const float*)d_in[3];
    const float* w_res = (const float*)d_in[4];
    float* out = (float*)d_out;

    // ---- workspace carve (256B aligned) ----
    char* wp = (char*)d_ws;
    size_t off = 0;
    auto alloc = [&](size_t bytes) -> void* {
        void* p = wp + off;
        off += (bytes + 255) & ~(size_t)255;
        return p;
    };
    const long NQKV = (long)BATCH * SEQ * DIM;         // 16.7M
    const long NBHD = (long)BH * SEQ * DHEAD;          // 16.7M
    h16*   xh    = (h16*)alloc(NQKV * 2);
    h16*   wT    = (h16*)alloc((long)3 * DIM * DIM * 2);
    h16*   qh    = (h16*)alloc(NBHD * 2);
    h16*   kh    = (h16*)alloc(NBHD * 2);
    h16*   vh    = (h16*)alloc(NBHD * 2);
    h16*   vTh   = (h16*)alloc(NBHD * 2);
    h16*   qlh   = (h16*)alloc((long)BH * NLM * DHEAD * 2);
    h16*   klh   = (h16*)alloc((long)BH * NLM * DHEAD * 2);
    float* attn2 = (float*)alloc((long)BH * 65536 * 4);
    h16*   x2h   = (h16*)alloc((long)BH * 65536 * 2);
    h16*   a1h   = (h16*)alloc((long)BH * 65536 * 2);
    h16*   t1Th  = (h16*)alloc((long)BH * 65536 * 2);
    h16*   t2Th  = (h16*)alloc((long)BH * 65536 * 2);
    h16*   t3Th  = (h16*)alloc((long)BH * 65536 * 2);
    h16*   zhA   = (h16*)alloc((long)BH * 65536 * 2);
    h16*   zTA   = (h16*)alloc((long)BH * 65536 * 2);
    h16*   zhB   = (h16*)alloc((long)BH * 65536 * 2);
    h16*   zTB   = (h16*)alloc((long)BH * 65536 * 2);
    float* a3v   = (float*)alloc((long)BH * NLM * DHEAD * 4);
    h16*   a3vT  = (h16*)alloc((long)BH * NLM * DHEAD * 2);
    h16*   WTh   = (h16*)alloc((long)BH * DHEAD * NLM * 2);
    float* parts = (float*)alloc((long)BH * 64 * 64 * 4);
    unsigned* sc = (unsigned*)alloc(64 * 4);

    // ---- pipeline ----
    k_init_scalars<<<1, 64, 0, stream>>>(sc);
    k_convert_x<<<(int)((NQKV + 255) / 256), 256, 0, stream>>>(x, xh, NQKV);
    k_convert_wT<<<(3 * DIM * DIM + 255) / 256, 256, 0, stream>>>(w_qkv, wT);
    k_qkv_gemm<<<dim3(256, 24), 256, 0, stream>>>(xh, wT, qh, kh, vh, vTh);
    k_landmark_means<<<BH * NLM, 64, 0, stream>>>(qh, kh, qlh, klh);

    // sim2 = q_l @ k_l^T  (f32 out for softmax/pinv scaling)
    k_gemm_f16<<<dim3(BH, 16, 16), 32, 0, stream>>>(
        attn2, nullptr, nullptr, qlh, klh,
        /*K*/64, /*lda*/64, /*ldb*/64, /*ldc*/256, /*ldct*/0,
        (long)NLM * DHEAD, (long)NLM * DHEAD, 65536, 0, 0.f, 1.f);
    k_softmax256<<<BH * 256, 256, 0, stream>>>(attn2, x2h);
    k_absmax<<<BH * 256, 256, 0, stream>>>(attn2, sc);
    k_zinit<<<(BH * 65536 + 255) / 256, 256, 0, stream>>>(attn2, sc, zhA, zTA);

    // Moore-Penrose iterations, all WMMA with transposed-f16 chaining
    h16 *zc = zhA, *zTc = zTA, *zn = zhB, *zTn = zTB;
    for (int it = 0; it < 6; ++it) {
        // A1 = X @ Z
        k_gemm_f16<<<dim3(BH, 16, 16), 32, 0, stream>>>(
            nullptr, a1h, nullptr, x2h, zTc,
            256, 256, 256, 256, 0, 65536, 65536, 65536, 0, 0.f, 1.f);
        // T1^T = 7I - A1^T
        k_aI_minus_AT<<<(BH * 65536 + 255) / 256, 256, 0, stream>>>(a1h, t1Th, 7.f);
        // T2^T = 15I - (A1 @ T1)^T
        k_gemm_f16<<<dim3(BH, 16, 16), 32, 0, stream>>>(
            nullptr, nullptr, t2Th, a1h, t1Th,
            256, 256, 256, 0, 256, 65536, 65536, 0, 65536, 15.f, -1.f);
        // T3^T = 13I - (A1 @ T2)^T
        k_gemm_f16<<<dim3(BH, 16, 16), 32, 0, stream>>>(
            nullptr, nullptr, t3Th, a1h, t2Th,
            256, 256, 256, 0, 256, 65536, 65536, 0, 65536, 13.f, -1.f);
        // Z' = 0.25 * Z @ T3   (emit row-major and transposed f16)
        k_gemm_f16<<<dim3(BH, 16, 16), 32, 0, stream>>>(
            nullptr, zn, zTn, zc, t3Th,
            256, 256, 256, 256, 256, 65536, 65536, 65536, 65536, 0.f, 0.25f);
        h16* t;
        t = zc; zc = zn; zn = t;
        t = zTc; zTc = zTn; zTn = t;
    }

    // attn3v = softmax(q_l @ k^T) @ v   (fused, 128KB LDS logit block)
    k_sim3_softmax_pv<<<dim3(BH, 16), 256, 16 * SEQ * 2, stream>>>(qlh, kh, vTh, a3v);
    k_a3v_transpose<<<(BH * NLM * DHEAD + 255) / 256, 256, 0, stream>>>(a3v, a3vT);

    // W^T = (Z @ attn3v)^T  -> [b,h,64,256] f16
    k_gemm_f16<<<dim3(BH, 16, 4), 32, 0, stream>>>(
        nullptr, nullptr, WTh, zc, a3vT,
        256, 256, 256, 0, 256, 65536, (long)DHEAD * NLM, 0, (long)DHEAD * NLM, 0.f, 1.f);

    // out blocks = softmax(q @ k_l^T) @ W + depthwise conv residual, max-partials
    k_sim1_softmax_out<<<dim3(BH, 64), 256, 0, stream>>>(qh, klh, WTh, vh, w_res, parts);

    // final: max over sequence, project to dim
    k_final_proj<<<BH, 512, 0, stream>>>(parts, w_out, b_out, out);
}
